// GCNLayer_6064493821989
// MI455X (gfx1250) — matrix operand order repaired and checked by reading, not compiled, not run on hardware
//
#include <hip/hip_runtime.h>

typedef _Float16 v16h __attribute__((ext_vector_type(16)));
typedef _Float16 v8h  __attribute__((ext_vector_type(8)));
typedef float    v8f  __attribute__((ext_vector_type(8)));

#define WMMA_F16(a, b, c) \
  __builtin_amdgcn_wmma_f32_16x16x32_f16(false, (a), false, (b), (short)0, (c), false, false)

// ---------------------------------------------------------------------------
// CDNA5 async-tensor path helpers (ASYNCcnt + global_load_async_to_lds)
// ---------------------------------------------------------------------------
__device__ __forceinline__ unsigned lds_off_u32(const void* p) {
  // generic LDS address: low 32 bits == byte offset within LDS allocation
  return (unsigned)(unsigned long long)p;
}

__device__ __forceinline__ void async_ld_b128(unsigned lds, unsigned goff, const float* base) {
  // GVS mode: mem = SADDR + VADDR[31:0]; dst VGPR = LDS byte address
  asm volatile("global_load_async_to_lds_b128 %0, %1, %2"
               :: "v"(lds), "v"(goff), "s"(base) : "memory");
}
__device__ __forceinline__ void wait_async_le2() { asm volatile("s_wait_asynccnt 2" ::: "memory"); }
__device__ __forceinline__ void wait_async_le0() { asm volatile("s_wait_asynccnt 0" ::: "memory"); }

// ---------------------------------------------------------------------------
// fragment builders
// ---------------------------------------------------------------------------
__device__ __forceinline__ v16h cvt16_f32(const float* __restrict__ p) {
  v16h r;
#pragma unroll
  for (int k = 0; k < 16; ++k) r[k] = (_Float16)p[k];
  return r;
}

// A fragment: halves kb..kb+15 of a padded f16 LDS row
__device__ __forceinline__ v16h afrag_lds_h(const _Float16* __restrict__ a, int kb) {
  v8h lo = *(const v8h*)(a + kb);
  v8h hi = *(const v8h*)(a + kb + 8);
  return __builtin_shufflevector(lo, hi, 0, 1, 2, 3, 4, 5, 6, 7,
                                         8, 9, 10, 11, 12, 13, 14, 15);
}

// B fragment: 16 contiguous halves
__device__ __forceinline__ v16h bfrag_lds_h(const _Float16* __restrict__ p) {
  v8h lo = *(const v8h*)(p);
  v8h hi = *(const v8h*)(p + 8);
  return __builtin_shufflevector(lo, hi, 0, 1, 2, 3, 4, 5, 6, 7,
                                         8, 9, 10, 11, 12, 13, 14, 15);
}

// cooperative f32->f16 convert of 8 contiguous floats per thread
__device__ __forceinline__ void cvt8_store(const float* __restrict__ src,
                                           _Float16* __restrict__ dst) {
  float4 a  = *(const float4*)(src);
  float4 b4 = *(const float4*)(src + 4);
  v8h h;
  h[0] = (_Float16)a.x;  h[1] = (_Float16)a.y;
  h[2] = (_Float16)a.z;  h[3] = (_Float16)a.w;
  h[4] = (_Float16)b4.x; h[5] = (_Float16)b4.y;
  h[6] = (_Float16)b4.z; h[7] = (_Float16)b4.w;
  *(v8h*)dst = h;
}

// ---------------------------------------------------------------------------
// K0: node GEMMs  out[n,c] = sum_h x[n,h]*W[c,h] + b[c]   (2048 x 128, x3)
// ---------------------------------------------------------------------------
__global__ __launch_bounds__(256)
void node_gemm(const float* __restrict__ x,
               const float* __restrict__ W0, const float* __restrict__ B0, float* __restrict__ O0,
               const float* __restrict__ W1, const float* __restrict__ B1, float* __restrict__ O1,
               const float* __restrict__ W2, const float* __restrict__ B2, float* __restrict__ O2) {
  __shared__ _Float16 sXh[16 * 136];
  const int tile = blockIdx.x;   // 0..127 (16 rows each)
  const int mat  = blockIdx.y;   // 0..2
  const float* W  = (mat == 0) ? W0 : ((mat == 1) ? W1 : W2);
  const float* Bb = (mat == 0) ? B0 : ((mat == 1) ? B1 : B2);
  float*       O  = (mat == 0) ? O0 : ((mat == 1) ? O1 : O2);
  const int t = threadIdx.x;
  cvt8_store(x + (size_t)tile * 2048 + t * 8, sXh + (t >> 4) * 136 + (t & 15) * 8);
  __syncthreads();
  const int w = t >> 5, lane = t & 31, lh = lane >> 4, l15 = lane & 15, c = w * 16 + l15;
  v8f acc = {};
  const _Float16* arow = sXh + l15 * 136;
#pragma unroll
  for (int s = 0; s < 4; ++s) {
    v16h af = afrag_lds_h(arow, s * 32 + lh * 16);
    v16h bf = cvt16_f32(W + c * 128 + s * 32 + lh * 16);   // B[k][c] = W[c][k]
    acc = WMMA_F16(af, bf, acc);
  }
  const float bv = Bb[c];
  float* orow = O + (size_t)tile * 2048 + (size_t)(lh * 8) * 128 + c;
#pragma unroll
  for (int g = 0; g < 8; ++g) orow[(size_t)g * 128] = acc[g] + bv;
}

// ---------------------------------------------------------------------------
// K1/K4: edge GEMM passes. One workgroup (8 waves) per (b,i); 16 j-tiles of
// 16 rows. Pipeline: async global->LDS f32 (double buffered) -> cooperative
// f32->f16 LDS convert (double buffered) -> WMMA with A and B both from LDS
// f16 (weights staged once per workgroup).
// PASS==1: BN stats + gated aggregation.  PASS==2: fold-BN, ReLU, residual.
// ---------------------------------------------------------------------------
template <int PASS>
__global__ __launch_bounds__(256)
void edge_pass(const float* __restrict__ e, const float* __restrict__ Uw,
               const float* __restrict__ Ub, const float* __restrict__ Vx,
               const float* __restrict__ Vnx,
               float* __restrict__ agg, float* __restrict__ esum, float* __restrict__ esumsq,
               const float* __restrict__ escale, const float* __restrict__ eshift,
               float* __restrict__ eout) {
  __shared__ float    sF[2][16 * 132];   // raw f32 tiles (async dst; residual in pass2)
  __shared__ _Float16 sH[2][16 * 136];   // converted f16 tiles (WMMA A source)
  __shared__ _Float16 sW[128 * 136];     // f16 weights (WMMA B source)
  const int bi   = blockIdx.x;           // b*256 + i
  const int b    = bi >> 8;
  const int t    = threadIdx.x;
  const int w    = t >> 5;
  const int lane = t & 31;
  const int lh   = lane >> 4;
  const int l15  = lane & 15;
  const int c    = w * 16 + l15;         // this lane's output channel

  // staging geometry: thread t owns 32B of the 8KB tile (two b128 transfers)
  const float*   tbase = e + (size_t)bi * 32768;          // e[b,i,:,:]
  const int      rr    = t >> 4;
  const int      qq    = t & 15;
  const unsigned loff  = (unsigned)(rr * 528 + qq * 32);  // padded f32 LDS row (132 fl)
  const unsigned goff  = (unsigned)(rr * 512 + qq * 32);
  const unsigned lds0  = lds_off_u32(&sF[0][0]);
  const unsigned lds1  = lds_off_u32(&sF[1][0]);
  const float*    cvs0 = &sF[0][rr * 132 + qq * 8];       // convert-stage src
  const float*    cvs1 = &sF[1][rr * 132 + qq * 8];
  _Float16*       cvd0 = &sH[0][rr * 136 + qq * 8];       // convert-stage dst
  _Float16*       cvd1 = &sH[1][rr * 136 + qq * 8];

  // kick off tiles 0 and 1 via the TDM-style async path
  async_ld_b128(lds0 + loff,      goff,             tbase);
  async_ld_b128(lds0 + loff + 16, goff + 16,        tbase);
  async_ld_b128(lds1 + loff,      goff + 8192,      tbase);
  async_ld_b128(lds1 + loff + 16, goff + 8192 + 16, tbase);

  // stage weights to f16 LDS while the async transfers fly:
  // thread t converts 64 floats of row (t>>1), half (t&1)
  {
    const int wr = t >> 1, whalf = (t & 1) * 64;
#pragma unroll
    for (int k = 0; k < 8; ++k)
      cvt8_store(Uw + (size_t)wr * 128 + whalf + k * 8, sW + wr * 136 + whalf + k * 8);
  }

  const float bias_i = Ub[c] + Vx[(size_t)bi * 128 + c];  // Ue_b[c] + Vx[b,i,c]
  float esc = 0.f, esh = 0.f;
  if (PASS == 2) { esc = escale[c]; esh = eshift[c]; }

  wait_async_le2();   // tile 0 complete (tile 1 in flight)
  __syncthreads();
  cvt8_store(cvs0, cvd0);   // convert tile 0
  __syncthreads();

  float aggc = 0.f, ssum = 0.f, ssq = 0.f;

  for (int jt = 0; jt < 16; ++jt) {
    const int cur = jt & 1;
    // ---- compute: A and B fragments both from f16 LDS ----
    const _Float16* arow = &sH[cur][l15 * 136];
    const _Float16* brow = &sW[c * 136];
    v8f acc = {};
#pragma unroll
    for (int s = 0; s < 4; ++s) {
      v16h af = afrag_lds_h(arow, s * 32 + lh * 16);
      v16h bf = bfrag_lds_h(brow + s * 32 + lh * 16);
      acc = WMMA_F16(af, bf, acc);
    }
    const int    j0  = jt * 16 + lh * 8;                      // first j row for this lane
    const float* vxj = Vx + (size_t)b * 32768 + (size_t)j0 * 128 + c;
    if (PASS == 1) {
      const float* vnj = Vnx + (size_t)b * 32768 + (size_t)j0 * 128 + c;
#pragma unroll
      for (int g = 0; g < 8; ++g) {
        float v = acc[g] + bias_i + vxj[(size_t)g * 128];     // e_tmp
        ssum += v; ssq += v * v;                              // edge BN stats
        float gate = 1.0f / (1.0f + __expf(-v));              // sigmoid
        aggc += gate * vnj[(size_t)g * 128];                  // sum_j gate*Vnx
      }
    } else {
      float*       orow  = eout + (size_t)bi * 32768 + (size_t)j0 * 128 + c;
      const float* eorig = &sF[cur][(lh * 8) * 132 + c];      // original e from LDS
#pragma unroll
      for (int g = 0; g < 8; ++g) {
        float v   = acc[g] + bias_i + vxj[(size_t)g * 128];
        float bnv = fmaf(v, esc, esh);
        bnv = bnv > 0.f ? bnv : 0.f;
        __builtin_nontemporal_store(eorig[g * 132] + bnv, orow + (size_t)g * 128);
      }
    }
    __syncthreads();                       // all waves done with sF[cur] & sH[cur]
    // ---- refill sF[cur] with tile jt+2; ensure tile jt+1 resident ----
    if (jt + 2 < 16) {
      const unsigned dst = (cur == 0) ? lds0 : lds1;
      const unsigned go  = goff + (unsigned)(jt + 2) * 8192u;
      async_ld_b128(dst + loff,      go,      tbase);
      async_ld_b128(dst + loff + 16, go + 16, tbase);
      wait_async_le2();                    // tile jt+1 complete
    } else {
      wait_async_le0();
    }
    __syncthreads();                       // sF[nxt] visible to all waves
    // ---- convert tile jt+1 to f16 ----
    if (jt + 1 < 16) {
      cvt8_store((cur == 0) ? cvs1 : cvs0, (cur == 0) ? cvd1 : cvd0);
      __syncthreads();                     // sH[nxt] ready
    }
  }

  if (PASS == 1) {
    // lanes L and L+16 hold the same channel c over disjoint row halves
    ssum += __shfl_xor(ssum, 16);
    ssq  += __shfl_xor(ssq,  16);
    aggc += __shfl_xor(aggc, 16);
    if (lh == 0) {
      agg[(size_t)bi * 128 + c] = aggc;    // exclusive per (b,i): plain store
      atomicAdd(&esum[c],   ssum);
      atomicAdd(&esumsq[c], ssq);
    }
  }
}

// ---------------------------------------------------------------------------
// K2: fold BN params; build x_tmp = Ux + agg and its per-channel stats
// ---------------------------------------------------------------------------
__global__ void finalize_stats(const float* __restrict__ esum, const float* __restrict__ esumsq,
                               const float* __restrict__ eg, const float* __restrict__ ebp,
                               float* __restrict__ escale, float* __restrict__ eshift,
                               const float* __restrict__ Ux, const float* __restrict__ agg,
                               float* __restrict__ xtmp,
                               const float* __restrict__ ng, const float* __restrict__ nbp,
                               float* __restrict__ nscale, float* __restrict__ nshift) {
  const int c = threadIdx.x;  // 128 threads
  {
    const float Me  = 524288.0f;   // B*N*N
    float m   = esum[c] / Me;
    float var = esumsq[c] / Me - m * m;
    float sc  = eg[c] * rsqrtf(var + 1e-5f);
    escale[c] = sc;
    eshift[c] = ebp[c] - m * sc;
  }
  float s = 0.f, s2 = 0.f;
  for (int r = 0; r < 2048; ++r) {
    float v = Ux[(size_t)r * 128 + c] + agg[(size_t)r * 128 + c];
    xtmp[(size_t)r * 128 + c] = v;
    s += v; s2 += v * v;
  }
  float mn = s / 2048.f;
  float vr = s2 / 2048.f - mn * mn;
  float sn = ng[c] * rsqrtf(vr + 1e-5f);
  nscale[c] = sn;
  nshift[c] = nbp[c] - mn * sn;
}

// ---------------------------------------------------------------------------
// K3: x_new = x + relu(bn(x_tmp))
// ---------------------------------------------------------------------------
__global__ __launch_bounds__(256)
void node_out(const float* __restrict__ x, const float* __restrict__ xtmp,
              const float* __restrict__ nscale, const float* __restrict__ nshift,
              float* __restrict__ out) {
  const int idx = blockIdx.x * 256 + threadIdx.x;
  const int c   = idx & 127;
  float v = fmaf(xtmp[idx], nscale[c], nshift[c]);
  out[idx] = x[idx] + (v > 0.f ? v : 0.f);
}

// ---------------------------------------------------------------------------
extern "C" void kernel_launch(void* const* d_in, const int* in_sizes, int n_in,
                              void* d_out, int out_size, void* d_ws, size_t ws_size,
                              hipStream_t stream) {
  (void)in_sizes; (void)n_in; (void)out_size; (void)ws_size;

  const float* x    = (const float*)d_in[0];
  const float* e    = (const float*)d_in[1];
  const float* Ue_w = (const float*)d_in[2];
  const float* Ue_b = (const float*)d_in[3];
  const float* Ve_w = (const float*)d_in[4];
  const float* Ve_b = (const float*)d_in[5];
  const float* Un_w = (const float*)d_in[6];
  const float* Un_b = (const float*)d_in[7];
  const float* Vn_w = (const float*)d_in[8];
  const float* Vn_b = (const float*)d_in[9];
  const float* ng   = (const float*)d_in[10];
  const float* nb   = (const float*)d_in[11];
  const float* eg   = (const float*)d_in[12];
  const float* eb   = (const float*)d_in[13];

  float* ws     = (float*)d_ws;
  float* Vx     = ws;                 // 262144
  float* Ux     = ws + 262144;        // 262144
  float* Vnx    = ws + 524288;        // 262144
  float* agg    = ws + 786432;        // 262144
  float* xtmp   = ws + 1048576;       // 262144
  float* esum   = ws + 1310720;       // 128
  float* esumsq = esum + 128;
  float* escale = esum + 256;
  float* eshift = esum + 384;
  float* nscale = esum + 512;
  float* nshift = esum + 640;

  float* xout = (float*)d_out;        // [B,N,H]
  float* eout = xout + 262144;        // [B,N,N,H]

  hipMemsetAsync(esum, 0, 256 * sizeof(float), stream);

  node_gemm<<<dim3(128, 3), 256, 0, stream>>>(x, Ve_w, Ve_b, Vx, Un_w, Un_b, Ux, Vn_w, Vn_b, Vnx);

  edge_pass<1><<<2048, 256, 0, stream>>>(e, Ue_w, Ue_b, Vx, Vnx,
                                         agg, esum, esumsq, nullptr, nullptr, nullptr);

  finalize_stats<<<1, 128, 0, stream>>>(esum, esumsq, eg, eb, escale, eshift,
                                        Ux, agg, xtmp, ng, nb, nscale, nshift);

  node_out<<<1024, 256, 0, stream>>>(x, xtmp, nscale, nshift, xout);

  edge_pass<2><<<2048, 256, 0, stream>>>(e, Ue_w, Ue_b, Vx, nullptr,
                                         nullptr, nullptr, nullptr, escale, eshift, eout);
}
